// GIN_30580167148116
// MI455X (gfx1250) — matrix operand order repaired and checked by reading, not compiled, not run on hardware
//
#include <hip/hip_runtime.h>
#include <hip/hip_bf16.h>

#define NNODES 100000
#define NEDGES 800000
#define INC 128
#define HID 512
#define OUTC 2
#define NGRAPHS 512
#define BN_EPS 1e-5f

typedef __attribute__((ext_vector_type(16))) _Float16 v16h;
typedef __attribute__((ext_vector_type(8)))  _Float16 h8v;
typedef __attribute__((ext_vector_type(4)))  _Float16 h4v;
typedef __attribute__((ext_vector_type(8)))  float    v8f;

union Frag { v16h v; h8v h[2]; };

// ---------------- edge scatter: hsum[dst] += h[src], float4 per thread ----------------
__global__ void k_scatter(const float* __restrict__ h, const int* __restrict__ ei,
                          float* __restrict__ hsum, int E, int C, int qshift) {
  long long idx = (long long)blockIdx.x * blockDim.x + threadIdx.x;
  int quads = C >> 2;
  long long total = (long long)E * quads;
  if (idx >= total) return;
  int e = (int)(idx >> qshift);
  int q = (int)(idx & (quads - 1));
  int s = ei[e], d = ei[E + e];
  float4 v = *(const float4*)(h + (size_t)s * C + q * 4);
  float* p = hsum + (size_t)d * C + q * 4;
  atomicAdd(p + 0, v.x);
  atomicAdd(p + 1, v.y);
  atomicAdd(p + 2, v.z);
  atomicAdd(p + 3, v.w);
}

// ---------------- f32 -> f16 (4 elems / thread) ----------------
__global__ void k_cvt_half4(const float* __restrict__ in, _Float16* __restrict__ out, long long n4) {
  long long i = (long long)blockIdx.x * blockDim.x + threadIdx.x;
  if (i >= n4) return;
  float4 v = ((const float4*)in)[i];
  h4v o;
  o[0] = (_Float16)v.x; o[1] = (_Float16)v.y; o[2] = (_Float16)v.z; o[3] = (_Float16)v.w;
  ((h4v*)out)[i] = o;
}

// ---------------- weight convert+transpose: W[K][N] f32 -> Wt[N][K] f16 ----------------
__global__ void k_wt(const float* __restrict__ W, _Float16* __restrict__ Wt, int K, int N) {
  int idx = blockIdx.x * blockDim.x + threadIdx.x;
  if (idx >= K * N) return;
  int k = idx / N;
  int n = idx - k * N;
  Wt[(size_t)n * K + k] = (_Float16)W[idx];
}

// ---------------- WMMA GEMM: out = relu(A[MxK] @ Wt[NxK]^T + bias) ----------------
// block = 256 threads = 8 waves; wave tile = 16(M) x 64(N); block tile = 128 x 64
__global__ __launch_bounds__(256) void k_gemm(const _Float16* __restrict__ A,
                                              const _Float16* __restrict__ Wt,
                                              const float* __restrict__ bias,
                                              int M, int N, int K,
                                              float* __restrict__ outF,
                                              _Float16* __restrict__ outH,
                                              int relu) {
  int lane = threadIdx.x & 31;
  int wave = threadIdx.x >> 5;
  int mbase = blockIdx.x * 128 + wave * 16;
  int nbase = blockIdx.y * 64;
  int lrow = mbase + (lane & 15);
  if (lrow >= M) lrow = M - 1;           // clamp loads; garbage rows never stored
  int hi = lane >> 4;                     // lane group 0/1
  const _Float16* arow = A + (size_t)lrow * K;

  v8f acc0 = {}, acc1 = {}, acc2 = {}, acc3 = {};
  for (int kb = 0; kb < K; kb += 32) {
    Frag a;
    // A k order used here (contiguous, matched by B; not the ISA table order): lanes 0-15 hold K 0..15, lanes 16-31 hold K 16..31
    a.h[0] = *(const h8v*)(arow + kb + hi * 16);
    a.h[1] = *(const h8v*)(arow + kb + hi * 16 + 8);
    Frag b0, b1, b2, b3;
    // B k order used here (contiguous, same as A): lanes 0-15 hold K=0..15 of col n, lanes 16-31 hold K=16..31
    const _Float16* bc0 = Wt + (size_t)(nbase +  0 + (lane & 15)) * K + kb + hi * 16;
    const _Float16* bc1 = Wt + (size_t)(nbase + 16 + (lane & 15)) * K + kb + hi * 16;
    const _Float16* bc2 = Wt + (size_t)(nbase + 32 + (lane & 15)) * K + kb + hi * 16;
    const _Float16* bc3 = Wt + (size_t)(nbase + 48 + (lane & 15)) * K + kb + hi * 16;
    b0.h[0] = *(const h8v*)(bc0); b0.h[1] = *(const h8v*)(bc0 + 8);
    b1.h[0] = *(const h8v*)(bc1); b1.h[1] = *(const h8v*)(bc1 + 8);
    b2.h[0] = *(const h8v*)(bc2); b2.h[1] = *(const h8v*)(bc2 + 8);
    b3.h[0] = *(const h8v*)(bc3); b3.h[1] = *(const h8v*)(bc3 + 8);
    acc0 = __builtin_amdgcn_wmma_f32_16x16x32_f16(false, a.v, false, b0.v, (short)0, acc0, false, false);
    acc1 = __builtin_amdgcn_wmma_f32_16x16x32_f16(false, a.v, false, b1.v, (short)0, acc1, false, false);
    acc2 = __builtin_amdgcn_wmma_f32_16x16x32_f16(false, a.v, false, b2.v, (short)0, acc2, false, false);
    acc3 = __builtin_amdgcn_wmma_f32_16x16x32_f16(false, a.v, false, b3.v, (short)0, acc3, false, false);
  }

  v8f accs[4] = {acc0, acc1, acc2, acc3};
#pragma unroll
  for (int j = 0; j < 4; ++j) {
    int col = nbase + j * 16 + (lane & 15);
    float bv = bias ? bias[col] : 0.0f;
#pragma unroll
    for (int r = 0; r < 8; ++r) {
      int m = mbase + r + hi * 8;        // D layout: VGPR r = row r (lanes 0-15) / r+8 (16-31)
      if (m < M) {
        float v = accs[j][r] + bv;
        if (relu) v = fmaxf(v, 0.0f);
        if (outF) outF[(size_t)m * N + col] = v;
        if (outH) outH[(size_t)m * N + col] = (_Float16)v;
      }
    }
  }
}

// ---------------- BN stats: per-feature sum / sumsq (C = 512), LDS partials ----------------
__global__ __launch_bounds__(256) void k_bn_partial(const float* __restrict__ h,
                                                    float* __restrict__ gsum,
                                                    float* __restrict__ gsq, long long n) {
  __shared__ float ls[HID];
  __shared__ float lq[HID];
  for (int i = threadIdx.x; i < HID; i += blockDim.x) { ls[i] = 0.f; lq[i] = 0.f; }
  __syncthreads();
  long long stride = (long long)gridDim.x * blockDim.x;
  for (long long i = (long long)blockIdx.x * blockDim.x + threadIdx.x; i < n; i += stride) {
    float v = h[i];
    int c = (int)(i & (HID - 1));
    atomicAdd(&ls[c], v);
    atomicAdd(&lq[c], v * v);
  }
  __syncthreads();
  for (int i = threadIdx.x; i < HID; i += blockDim.x) {
    atomicAdd(&gsum[i], ls[i]);
    atomicAdd(&gsq[i], lq[i]);
  }
}

__global__ void k_bn_fin(const float* sum, const float* sq, const float* g, const float* be,
                         float* scale, float* shift, float invM) {
  int c = threadIdx.x;
  float mean = sum[c] * invM;
  float var  = sq[c] * invM - mean * mean;
  float s = g[c] * rsqrtf(var + BN_EPS);
  scale[c] = s;
  shift[c] = be[c] - mean * s;
}

__global__ void k_bn_apply4(float* __restrict__ h, const float* __restrict__ scale,
                            const float* __restrict__ shift, long long n4) {
  long long i = (long long)blockIdx.x * blockDim.x + threadIdx.x;
  if (i >= n4) return;
  float4 v = ((float4*)h)[i];
  int c = (int)((i * 4) & (HID - 1));
  v.x = v.x * scale[c]     + shift[c];
  v.y = v.y * scale[c + 1] + shift[c + 1];
  v.z = v.z * scale[c + 2] + shift[c + 2];
  v.w = v.w * scale[c + 3] + shift[c + 3];
  ((float4*)h)[i] = v;
}

// ---------------- pooling ----------------
__global__ void k_count(const int* __restrict__ batch, float* __restrict__ cnt, int n) {
  int i = blockIdx.x * blockDim.x + threadIdx.x;
  if (i < n) atomicAdd(&cnt[batch[i]], 1.0f);
}

__global__ void k_poolsum(const float* __restrict__ h, const int* __restrict__ batch,
                          float* __restrict__ gsum, long long total) {
  long long idx = (long long)blockIdx.x * blockDim.x + threadIdx.x;
  if (idx >= total) return;
  int node = (int)(idx >> 7);            // 128 quads per node (HID/4)
  int q = (int)(idx & 127);
  int g = batch[node];
  float4 v = *(const float4*)(h + (size_t)node * HID + q * 4);
  float* p = gsum + (size_t)g * HID + q * 4;
  atomicAdd(p + 0, v.x);
  atomicAdd(p + 1, v.y);
  atomicAdd(p + 2, v.z);
  atomicAdd(p + 3, v.w);
}

__global__ void k_pooldiv(const float* __restrict__ gsum, const float* __restrict__ cnt,
                          _Float16* __restrict__ poolh, int n) {
  int i = blockIdx.x * blockDim.x + threadIdx.x;
  if (i >= n) return;
  poolh[i] = (_Float16)(gsum[i] / fmaxf(cnt[i >> 9], 1.0f));
}

// ---------------- head final linear (HID -> 2) ----------------
__global__ void k_headlin(const float* __restrict__ z, const float* __restrict__ Wb,
                          const float* __restrict__ bb, float* __restrict__ last) {
  int idx = blockIdx.x * blockDim.x + threadIdx.x;
  if (idx >= NGRAPHS * OUTC) return;
  int g = idx >> 1, o = idx & 1;
  float acc = bb[o];
  const float* zr = z + (size_t)g * HID;
  for (int k = 0; k < HID; ++k) acc += zr[k] * Wb[k * OUTC + o];
  last[idx] = acc;
}

__global__ void k_softmax2(const float* __restrict__ last, float* __restrict__ logp,
                           float* __restrict__ soft) {
  int g = blockIdx.x * blockDim.x + threadIdx.x;
  if (g >= NGRAPHS) return;
  float a = last[2 * g], b = last[2 * g + 1];
  float m = fmaxf(a, b);
  float lse = m + logf(expf(a - m) + expf(b - m));
  float l0 = a - lse, l1 = b - lse;
  logp[2 * g] = l0; logp[2 * g + 1] = l1;
  float e0 = expf(l0), e1 = expf(l1), s = e0 + e1;
  soft[2 * g] = e0 / s; soft[2 * g + 1] = e1 / s;
}

// =====================================================================================
extern "C" void kernel_launch(void* const* d_in, const int* in_sizes, int n_in,
                              void* d_out, int out_size, void* d_ws, size_t ws_size,
                              hipStream_t stream) {
  const float* x     = (const float*)d_in[0];
  const int*   ei    = (const int*)d_in[1];
  const int*   batch = (const int*)d_in[3];
  const float* W1_0  = (const float*)d_in[4];
  const float* b1_0  = (const float*)d_in[5];
  const float* W2_0  = (const float*)d_in[6];
  const float* b2_0  = (const float*)d_in[7];
  const float* g_0   = (const float*)d_in[8];
  const float* be_0  = (const float*)d_in[9];
  const float* Ws1   = (const float*)d_in[10];
  const float* bs1   = (const float*)d_in[11];
  const float* Ws2   = (const float*)d_in[12];
  const float* bs2   = (const float*)d_in[13];
  const float* gs    = (const float*)d_in[14];
  const float* bes   = (const float*)d_in[15];
  const float* Wa    = (const float*)d_in[16];
  const float* ba    = (const float*)d_in[17];
  const float* ga    = (const float*)d_in[18];
  const float* bea   = (const float*)d_in[19];
  const float* Wb    = (const float*)d_in[20];
  const float* bb    = (const float*)d_in[21];

  // ---- workspace carve-up ----
  char* ws = (char*)d_ws;
  size_t off = 0;
  auto carve = [&](size_t bytes) -> void* {
    void* p = ws + off;
    off += (bytes + 255) & ~(size_t)255;
    return p;
  };
  float*    act   = (float*)carve((size_t)NNODES * HID * 4);     // activations (f32)
  float*    hsum  = (float*)carve((size_t)NNODES * HID * 4);     // x + agg (f32)
  _Float16* Ah    = (_Float16*)carve((size_t)NNODES * HID * 2);  // GEMM1 input (f16)
  _Float16* h1h   = (_Float16*)carve((size_t)NNODES * HID * 2);  // GEMM2 input (f16)
  _Float16* wbuf  = (_Float16*)carve((size_t)(HID * INC + 8 * HID * HID) * 2);
  float* bnsum    = (float*)carve(HID * 4);
  float* bnsq     = (float*)carve(HID * 4);
  float* bnscale  = (float*)carve(HID * 4);
  float* bnshift  = (float*)carve(HID * 4);
  float* gsum     = (float*)carve((size_t)NGRAPHS * HID * 4);
  float* gcnt     = (float*)carve(NGRAPHS * 4);
  _Float16* poolh = (_Float16*)carve((size_t)NGRAPHS * HID * 2);
  float* z        = (float*)carve((size_t)NGRAPHS * HID * 4);

  _Float16* W1t  = wbuf;                       // [HID][INC]
  _Float16* W2t  = W1t + HID * INC;            // [HID][HID]
  _Float16* Ws1t = W2t + HID * HID;            // 3 x [HID][HID]
  _Float16* Ws2t = Ws1t + 3 * HID * HID;       // 3 x [HID][HID]
  _Float16* Wat  = Ws2t + 3 * HID * HID;       // [HID][HID]

  // ---- weight conversion (per-call, deterministic) ----
  k_wt<<<(INC * HID + 255) / 256, 256, 0, stream>>>(W1_0, W1t, INC, HID);
  k_wt<<<(HID * HID + 255) / 256, 256, 0, stream>>>(W2_0, W2t, HID, HID);
  for (int i = 0; i < 3; ++i) {
    k_wt<<<(HID * HID + 255) / 256, 256, 0, stream>>>(Ws1 + (size_t)i * HID * HID, Ws1t + (size_t)i * HID * HID, HID, HID);
    k_wt<<<(HID * HID + 255) / 256, 256, 0, stream>>>(Ws2 + (size_t)i * HID * HID, Ws2t + (size_t)i * HID * HID, HID, HID);
  }
  k_wt<<<(HID * HID + 255) / 256, 256, 0, stream>>>(Wa, Wat, HID, HID);

  dim3 gemmGrid((NNODES + 127) / 128, HID / 64);

  auto run_layer = [&](const float* hin, int C, const _Float16* w1t, const float* b1,
                       const _Float16* w2t, const float* b2, const float* g, const float* be) {
    // hsum = hin + segment_sum(hin[src] -> dst)
    hipMemcpyAsync(hsum, hin, (size_t)NNODES * C * 4, hipMemcpyDeviceToDevice, stream);
    int qshift = (C == HID) ? 7 : 5;
    long long tot = (long long)NEDGES * (C >> 2);
    k_scatter<<<(int)((tot + 255) / 256), 256, 0, stream>>>(hin, ei, hsum, NEDGES, C, qshift);
    long long n4 = (long long)NNODES * C / 4;
    k_cvt_half4<<<(int)((n4 + 255) / 256), 256, 0, stream>>>(hsum, Ah, n4);
    // h1 = relu(hsum @ W1 + b1)  (f16 out for next GEMM)
    k_gemm<<<gemmGrid, 256, 0, stream>>>(Ah, w1t, b1, NNODES, HID, C, nullptr, h1h, 1);
    // h2 = relu(h1 @ W2 + b2)    (f32 out, overwrites act)
    k_gemm<<<gemmGrid, 256, 0, stream>>>(h1h, w2t, b2, NNODES, HID, HID, act, nullptr, 1);
    // BN (training-mode biased stats)
    hipMemsetAsync(bnsum, 0, HID * 4, stream);
    hipMemsetAsync(bnsq, 0, HID * 4, stream);
    k_bn_partial<<<512, 256, 0, stream>>>(act, bnsum, bnsq, (long long)NNODES * HID);
    k_bn_fin<<<1, HID, 0, stream>>>(bnsum, bnsq, g, be, bnscale, bnshift, 1.0f / (float)NNODES);
    long long a4 = (long long)NNODES * HID / 4;
    k_bn_apply4<<<(int)((a4 + 255) / 256), 256, 0, stream>>>(act, bnscale, bnshift, a4);
  };

  // layer 0 (K=128) then layers 1..3 (K=512)
  run_layer(x, INC, W1t, b1_0, W2t, b2_0, g_0, be_0);
  for (int i = 0; i < 3; ++i) {
    run_layer(act, HID, Ws1t + (size_t)i * HID * HID, bs1 + (size_t)i * HID,
              Ws2t + (size_t)i * HID * HID, bs2 + (size_t)i * HID,
              gs + (size_t)i * HID, bes + (size_t)i * HID);
  }

  // ---- global mean pool ----
  hipMemsetAsync(gsum, 0, (size_t)NGRAPHS * HID * 4, stream);
  hipMemsetAsync(gcnt, 0, NGRAPHS * 4, stream);
  k_count<<<(NNODES + 255) / 256, 256, 0, stream>>>(batch, gcnt, NNODES);
  long long pt = (long long)NNODES * (HID / 4);
  k_poolsum<<<(int)((pt + 255) / 256), 256, 0, stream>>>(act, batch, gsum, pt);
  k_pooldiv<<<(NGRAPHS * HID + 255) / 256, 256, 0, stream>>>(gsum, gcnt, poolh, NGRAPHS * HID);

  // ---- head: z = BN(relu(pooled @ Wa + ba)) ----
  dim3 headGrid((NGRAPHS + 127) / 128, HID / 64);
  k_gemm<<<headGrid, 256, 0, stream>>>(poolh, Wat, ba, NGRAPHS, HID, HID, z, nullptr, 1);
  hipMemsetAsync(bnsum, 0, HID * 4, stream);
  hipMemsetAsync(bnsq, 0, HID * 4, stream);
  k_bn_partial<<<128, 256, 0, stream>>>(z, bnsum, bnsq, (long long)NGRAPHS * HID);
  k_bn_fin<<<1, HID, 0, stream>>>(bnsum, bnsq, ga, bea, bnscale, bnshift, 1.0f / (float)NGRAPHS);
  long long z4 = (long long)NGRAPHS * HID / 4;
  k_bn_apply4<<<(int)((z4 + 255) / 256), 256, 0, stream>>>(z, bnscale, bnshift, z4);

  // ---- final linear + log_softmax / softmax ----
  float* out  = (float*)d_out;
  float* logp = out;                       // [512,2]
  float* soft = out + NGRAPHS * OUTC;      // [512,2]
  float* last = out + 2 * NGRAPHS * OUTC;  // [512,2]
  k_headlin<<<(NGRAPHS * OUTC + 255) / 256, 256, 0, stream>>>(z, Wb, bb, last);
  k_softmax2<<<(NGRAPHS + 255) / 256, 256, 0, stream>>>(last, logp, soft);
}
